// DecoderRNN_22187801051665
// MI455X (gfx1250) — compile-verified, matrix operand order checked by reading, not run on hardware
//
#include <hip/hip_runtime.h>
#include <cstdint>

// ---------------------------------------------------------------------------
// DecoderRNN on MI455X (gfx1250), wave32, bf16 WMMA.
// B=8192, H=1024, 3H=3072, H/2=512, IN=OUT=4, T_past=60, T_future=30.
// ---------------------------------------------------------------------------

typedef unsigned short u16;
typedef __attribute__((ext_vector_type(16))) __bf16 v16bf;
typedef __attribute__((ext_vector_type(8)))  float  v8f;

union FragAB { uint4 q[2]; v16bf v; };   // 16 bf16 = 32B per lane (A 16x32 / B 32x16 fragment)
union FragC  { v8f v; float f[8]; };     // 16x16 f32 C/D tile, 8 VGPRs per lane

#define Bsz 8192
#define Hd  1024
#define Gd  3072      // 3*H
#define H2  512
#define HP  1032      // LDS row pitch in halves (pad 8 to break bank alignment)
#define BM  64        // batch rows per persistent block

__device__ __forceinline__ float bf2f(u16 u) {
    uint32_t x = (uint32_t)u << 16; float f; __builtin_memcpy(&f, &x, 4); return f;
}
__device__ __forceinline__ u16 f2bf(float f) {
    uint32_t x; __builtin_memcpy(&x, &f, 4);
    x = (x + 0x7FFFu + ((x >> 16) & 1u)) >> 16;
    return (u16)x;
}
__device__ __forceinline__ float sigm(float x) { return 1.f / (1.f + __expf(-x)); }
__device__ __forceinline__ float tanh_fast(float x) {
    float e = __expf(-2.f * fabsf(x));
    float t = (1.f - e) / (1.f + e);
    return x < 0.f ? -t : t;
}

// ---------------------------------------------------------------------------
// fp32 -> bf16 conversion, 4 elements/thread
// ---------------------------------------------------------------------------
__global__ void cvt_f32_bf16_x4(const float* __restrict__ s, u16* __restrict__ d, int n4) {
    int i = blockIdx.x * blockDim.x + threadIdx.x;
    if (i < n4) {
        float4 v = ((const float4*)s)[i];
        ushort4 o;
        o.x = f2bf(v.x); o.y = f2bf(v.y); o.z = f2bf(v.z); o.w = f2bf(v.w);
        ((ushort4*)d)[i] = o;
    }
}

// ---------------------------------------------------------------------------
// Generic bf16 WMMA GEMM: C[M,N] = act(A[M,K] @ W[N,K]^T + bias[N]), C in bf16.
// Block = 256 threads = 8 waves, block tile 64(M) x 32(N), one 16x16 tile/wave.
// A,W fragments loaded straight from global (L2-resident working set).
// ---------------------------------------------------------------------------
__global__ void gemm_bf16_kernel(const u16* __restrict__ A, const u16* __restrict__ W,
                                 const float* __restrict__ bias, u16* __restrict__ C,
                                 int M, int N, int K, int act) {
    const int wave = threadIdx.x >> 5;
    const int lane = threadIdx.x & 31;
    const int l15  = lane & 15;
    const int kg   = lane >> 4;               // K-group select (0/1) per ISA A/B layout
    const int m0 = blockIdx.x * 64 + (wave & 3) * 16;
    const int n0 = blockIdx.y * 32 + (wave >> 2) * 16;

    FragC acc; acc.v = (v8f){0.f,0.f,0.f,0.f,0.f,0.f,0.f,0.f};

    const u16* arow = A + (size_t)(m0 + l15) * K;
    const u16* wrow = W + (size_t)(n0 + l15) * K;

    for (int kc = 0; kc < K; kc += 32) {
        FragAB a, b;
        const u16* ap = arow + kc + kg * 8;
        a.q[0] = *(const uint4*)(ap);
        a.q[1] = *(const uint4*)(ap + 16);
        const u16* wp = wrow + kc + kg * 8;
        b.q[0] = *(const uint4*)(wp);
        b.q[1] = *(const uint4*)(wp + 16);
        acc.v = __builtin_amdgcn_wmma_f32_16x16x32_bf16(false, a.v, false, b.v,
                                                        (short)0, acc.v, false, false);
    }

    const int j  = n0 + l15;
    const float bs = bias[j];
    const int rbase = m0 + (lane >= 16 ? 8 : 0);
#pragma unroll
    for (int r = 0; r < 8; ++r) {
        float v = acc.f[r] + bs;
        if (act) v = fmaxf(v, 0.f);
        C[(size_t)(rbase + r) * N + j] = f2bf(v);
    }
}

// ---------------------------------------------------------------------------
// Persistent fused GRU unroll + projection.
// Grid = 2 * (B/BM) blocks, 512 threads (16 waves).
// Blocks [0, B/BM)  -> GRU1 (T = past_length,  fc_in,  no relu)
// Blocks [B/BM, ..) -> GRU2 (T = future_length, fc_out, relu(h))
// h kept in LDS (bf16, double-buffered). wh streamed from L2 each step.
// ---------------------------------------------------------------------------
__global__ void gru_persistent_kernel(
    const u16* __restrict__ ctx_bf,
    const u16* __restrict__ gi1, const u16* __restrict__ wh1, const float* __restrict__ bh1,
    const float* __restrict__ fcw1, const float* __restrict__ fcb1,
    const u16* __restrict__ gi2, const u16* __restrict__ wh2, const float* __restrict__ bh2,
    const float* __restrict__ fcw2, const float* __restrict__ fcb2,
    float* __restrict__ out,
    const int* __restrict__ p_past, const int* __restrict__ p_future) {

    extern __shared__ u16 s_lds[];            // 2 * BM * HP halves = 258 KB

    const int T1 = *p_past;
    const int T2 = *p_future;
    const int half = gridDim.x >> 1;
    const int gru  = (blockIdx.x >= half) ? 1 : 0;
    const int blk  = gru ? (blockIdx.x - half) : blockIdx.x;
    const int m0   = blk * BM;

    const u16*   gi  = gru ? gi2  : gi1;
    const u16*   wh  = gru ? wh2  : wh1;
    const float* bh  = gru ? bh2  : bh1;
    const float* fcw = gru ? fcw2 : fcw1;
    const float* fcb = gru ? fcb2 : fcb1;
    const int    T   = gru ? T2 : T1;
    float* outbase   = out + (gru ? (size_t)Bsz * T1 * 4 : (size_t)0);

    const int tid  = threadIdx.x;
    const int wave = tid >> 5;
    const int lane = tid & 31;
    const int l15  = lane & 15;
    const int kg   = lane >> 4;

    // ---- load initial h = context rows [m0, m0+BM) into buffer 0 (bf16) ----
    {
        u16* h0 = s_lds;
        for (int idx = tid; idx < BM * (Hd / 8); idx += blockDim.x) {
            int row = idx >> 7;               // Hd/8 = 128 uint4 per row
            int c8  = idx & 127;
            uint4 v = *(const uint4*)(ctx_bf + (size_t)(m0 + row) * Hd + c8 * 8);
            *(uint4*)(h0 + row * HP + c8 * 8) = v;
        }
    }
    __syncthreads();

    for (int t = 0; t < T; ++t) {
        u16* cur = s_lds + (t & 1) * (BM * HP);
        u16* nxt = s_lds + ((t + 1) & 1) * (BM * HP);

        // ---- recurrent GEMM + gate math: 16 waves x 4 n-tiles each ----
#pragma unroll 1
        for (int q = 0; q < 4; ++q) {
            const int n0 = (wave * 4 + q) * 16;

            FragC acc[4][3];
#pragma unroll
            for (int mt = 0; mt < 4; ++mt)
#pragma unroll
                for (int g = 0; g < 3; ++g)
                    acc[mt][g].v = (v8f){0.f,0.f,0.f,0.f,0.f,0.f,0.f,0.f};

            const u16* wcol = wh + (size_t)(n0 + l15) * Hd;   // gate g adds g*Hd rows

#pragma unroll 1
            for (int kc = 0; kc < Hd; kc += 32) {
                FragAB bfr[3];
#pragma unroll
                for (int g = 0; g < 3; ++g) {
                    const u16* wp = wcol + (size_t)g * Hd * Hd + kc + kg * 8;
                    bfr[g].q[0] = *(const uint4*)(wp);
                    bfr[g].q[1] = *(const uint4*)(wp + 16);
                }
#pragma unroll
                for (int mt = 0; mt < 4; ++mt) {
                    FragAB a;
                    const u16* ap = cur + (mt * 16 + l15) * HP + kc + kg * 8;
                    a.q[0] = *(const uint4*)(ap);
                    a.q[1] = *(const uint4*)(ap + 16);
#pragma unroll
                    for (int g = 0; g < 3; ++g)
                        acc[mt][g].v = __builtin_amdgcn_wmma_f32_16x16x32_bf16(
                            false, a.v, false, bfr[g].v, (short)0, acc[mt][g].v, false, false);
                }
            }

            // ---- gate nonlinearity + h update -> nxt ----
            const int j = n0 + l15;
            const float bhr = bh[j];
            const float bhz = bh[Hd + j];
            const float bhn = bh[2 * Hd + j];
#pragma unroll
            for (int mt = 0; mt < 4; ++mt) {
#pragma unroll
                for (int r = 0; r < 8; ++r) {
                    const int row = mt * 16 + r + (lane >= 16 ? 8 : 0);
                    const u16* gp = gi + (size_t)(m0 + row) * Gd + j;
                    float ir = bf2f(gp[0]);
                    float iz = bf2f(gp[Hd]);
                    float in_ = bf2f(gp[2 * Hd]);
                    float rg = sigm(ir + acc[mt][0].f[r] + bhr);
                    float zg = sigm(iz + acc[mt][1].f[r] + bhz);
                    float ng = tanh_fast(in_ + rg * (acc[mt][2].f[r] + bhn));
                    float ho = bf2f(cur[row * HP + j]);
                    float hn = (1.f - zg) * ng + zg * ho;
                    nxt[row * HP + j] = f2bf(hn);
                }
            }
        }
        __syncthreads();

        // ---- projection: out[b, t, i] = sum_h h(relu?) * fcw[i,h] + fcb[i] ----
        // 256 (row, i) pairs; wave handles 16, lanes split H with stride 32.
#pragma unroll 1
        for (int p = wave * 16; p < wave * 16 + 16; ++p) {
            const int row = p >> 2;
            const int i   = p & 3;
            float s = 0.f;
#pragma unroll 4
            for (int jj = lane; jj < Hd; jj += 32) {
                float hv = bf2f(nxt[row * HP + jj]);
                if (gru) hv = fmaxf(hv, 0.f);
                s += hv * fcw[i * Hd + jj];
            }
#pragma unroll
            for (int off = 16; off > 0; off >>= 1) s += __shfl_xor(s, off, 32);
            if (lane == 0)
                outbase[((size_t)(m0 + row) * T + t) * 4 + i] = s + fcb[i];
        }
        __syncthreads();
    }
}

// ---------------------------------------------------------------------------
// Host-side launch
// ---------------------------------------------------------------------------
extern "C" void kernel_launch(void* const* d_in, const int* in_sizes, int n_in,
                              void* d_out, int out_size, void* d_ws, size_t ws_size,
                              hipStream_t stream) {
    const float* context = (const float*)d_in[0];
    const float* wc      = (const float*)d_in[1];
    const float* bc      = (const float*)d_in[2];
    const float* g1_wi   = (const float*)d_in[3];
    const float* g1_wh   = (const float*)d_in[4];
    const float* g1_bi   = (const float*)d_in[5];
    const float* g1_bh   = (const float*)d_in[6];
    const float* g2_wi   = (const float*)d_in[7];
    const float* g2_wh   = (const float*)d_in[8];
    const float* g2_bi   = (const float*)d_in[9];
    const float* g2_bh   = (const float*)d_in[10];
    const float* fc_in_w = (const float*)d_in[11];
    const float* fc_in_b = (const float*)d_in[12];
    const float* fc_out_w= (const float*)d_in[13];
    const float* fc_out_b= (const float*)d_in[14];
    const int*   p_fut   = (const int*)d_in[15];
    const int*   p_past  = (const int*)d_in[16];

    // workspace carve-out (bf16 buffers), 256B aligned
    char* ws = (char*)d_ws;
    size_t off = 0;
    auto carve = [&](size_t bytes) -> u16* {
        u16* p = (u16*)(ws + off);
        off += (bytes + 255) & ~(size_t)255;
        return p;
    };
    u16* ctx_bf = carve((size_t)Bsz * Hd * 2);   // 16 MB
    u16* ec_bf  = carve((size_t)Bsz * H2 * 2);   //  8 MB
    u16* gi1_bf = carve((size_t)Bsz * Gd * 2);   // 48 MB
    u16* gi2_bf = carve((size_t)Bsz * Gd * 2);   // 48 MB
    u16* wc_bf  = carve((size_t)H2  * Hd * 2);   //  1 MB
    u16* wi1_bf = carve((size_t)Gd  * H2 * 2);   //  3 MB
    u16* wi2_bf = carve((size_t)Gd  * H2 * 2);   //  3 MB
    u16* wh1_bf = carve((size_t)Gd  * Hd * 2);   //  6 MB
    u16* wh2_bf = carve((size_t)Gd  * Hd * 2);   //  6 MB

    auto cvt = [&](const float* s, u16* d, size_t n) {
        int n4 = (int)(n / 4);
        cvt_f32_bf16_x4<<<(n4 + 255) / 256, 256, 0, stream>>>(s, d, n4);
    };
    cvt(context, ctx_bf, (size_t)Bsz * Hd);
    cvt(wc,      wc_bf,  (size_t)H2  * Hd);
    cvt(g1_wi,   wi1_bf, (size_t)Gd  * H2);
    cvt(g1_wh,   wh1_bf, (size_t)Gd  * Hd);
    cvt(g2_wi,   wi2_bf, (size_t)Gd  * H2);
    cvt(g2_wh,   wh2_bf, (size_t)Gd  * Hd);

    // ec = relu(ctx @ wc^T + bc)       [8192 x 512], K=1024
    {
        dim3 g(Bsz / 64, H2 / 32);
        gemm_bf16_kernel<<<g, 256, 0, stream>>>(ctx_bf, wc_bf, bc, ec_bf, Bsz, H2, Hd, 1);
    }
    // gi = ec @ wi^T + bi              [8192 x 3072], K=512
    {
        dim3 g(Bsz / 64, Gd / 32);
        gemm_bf16_kernel<<<g, 256, 0, stream>>>(ec_bf, wi1_bf, g1_bi, gi1_bf, Bsz, Gd, H2, 0);
        gemm_bf16_kernel<<<g, 256, 0, stream>>>(ec_bf, wi2_bf, g2_bi, gi2_bf, Bsz, Gd, H2, 0);
    }

    // persistent fused GRU1 + GRU2 with in-block projection
    const size_t lds_bytes = (size_t)2 * BM * HP * sizeof(u16);   // 264192
    (void)hipFuncSetAttribute((const void*)gru_persistent_kernel,
                              hipFuncAttributeMaxDynamicSharedMemorySize, (int)lds_bytes);
    gru_persistent_kernel<<<2 * (Bsz / BM), 512, lds_bytes, stream>>>(
        ctx_bf,
        gi1_bf, wh1_bf, g1_bh, fc_in_w,  fc_in_b,
        gi2_bf, wh2_bf, g2_bh, fc_out_w, fc_out_b,
        (float*)d_out, p_past, p_fut);

    (void)in_sizes; (void)n_in; (void)out_size; (void)ws_size;
}
